// PointFeatureNet_57492432224357
// MI455X (gfx1250) — matrix operand order repaired and checked by reading, not compiled, not run on hardware
//
#include <hip/hip_runtime.h>

typedef _Float16 v8h  __attribute__((ext_vector_type(8)));
typedef _Float16 v16h __attribute__((ext_vector_type(16)));
typedef float    v8f  __attribute__((ext_vector_type(8)));

#define FBIG 3.0e38f

// ---------------------------------------------------------------------------
// WMMA fragment helpers (CDNA5 16x16x32 f16 -> f32)
// A (16x32 f16): lane<16 holds row m=lane, K {k0..k0+7}U{k0+16..k0+23};
// lane>=16 holds row m=lane-16, K {k0+8..15}U{k0+24..31}.  (ISA 7.12.2)
// B (32x16 f16) built from W^T rows: lane holds col n=lane&15,
// K same as A: k0 + 8*(lane>>4) + {0..7} U {16..23}.
// ---------------------------------------------------------------------------
__device__ __forceinline__ v16h cat16(v8h lo, v8h hi) {
  return __builtin_shufflevector(lo, hi, 0, 1, 2, 3, 4, 5, 6, 7,
                                 8, 9, 10, 11, 12, 13, 14, 15);
}

__device__ __forceinline__ v16h load_a_frag(const _Float16* buf, int strideH,
                                            int tile, int k0, int lane) {
  const int m  = lane & 15;
  const int hs = lane >> 4;
  const _Float16* p = buf + (tile * 16 + m) * strideH + k0 + hs * 8;
  v8h lo = *(const v8h*)(p);
  v8h hi = *(const v8h*)(p + 16);
  return cat16(lo, hi);
}

__device__ __forceinline__ v16h load_b_frag(const _Float16* wT, int strideH,
                                            int nt, int k0, int lane) {
  const int n  = lane & 15;
  const int hs = lane >> 4;
  const _Float16* p = wT + (nt * 16 + n) * strideH + k0 + hs * 8;
  v8h lo = *(const v8h*)(p);
  v8h hi = *(const v8h*)(p + 16);
  return cat16(lo, hi);
}

__device__ __forceinline__ v8f wmma_f16(v16h a, v16h b, v8f c) {
  return __builtin_amdgcn_wmma_f32_16x16x32_f16(
      /*neg_a=*/false, a, /*neg_b=*/false, b,
      /*c_mod=*/(short)0, c, /*reuse_a=*/false, /*reuse_b=*/false);
}

// Dense layer on one wave: in (32 x CINL, LDS) @ W (CINL x COUTL) + b, ReLU,
// write out (32 x COUTL, LDS). CINL, COUTL multiples of 32/16.
template <int CINL, int COUTL>
__device__ __forceinline__ void dense_relu_layer(const _Float16* inBuf,
                                                 const _Float16* wT,
                                                 const float* bias,
                                                 _Float16* outBuf, int lane) {
  const int n  = lane & 15;
  const int hs = lane >> 4;
  for (int nt = 0; nt < COUTL / 16; ++nt)
    for (int tile = 0; tile < 2; ++tile) {
      v8f acc = {};
      for (int ks = 0; ks < CINL / 32; ++ks) {
        v16h a = load_a_frag(inBuf, CINL, tile, ks * 32, lane);
        v16h w = load_b_frag(wT, CINL, nt, ks * 32, lane);
        acc = wmma_f16(a, w, acc);
      }
      float bb = bias[nt * 16 + n];
#pragma unroll
      for (int v = 0; v < 8; ++v) {
        float val = acc[v] + bb;
        val = val > 0.f ? val : 0.f;
        int r = tile * 16 + v + 8 * hs;
        outBuf[r * COUTL + nt * 16 + n] = (_Float16)val;
      }
    }
}

// ---------------------------------------------------------------------------
// split x (B,N,5) -> feat (B,N,2), pos (B,N,3)
// ---------------------------------------------------------------------------
__global__ void split_kernel(const float* __restrict__ x, float* __restrict__ feat,
                             float* __restrict__ pos, int total) {
  int i = blockIdx.x * blockDim.x + threadIdx.x;
  if (i >= total) return;
  pos[i * 3 + 0]  = x[i * 5 + 0];
  pos[i * 3 + 1]  = x[i * 5 + 1];
  pos[i * 3 + 2]  = x[i * 5 + 2];
  feat[i * 2 + 0] = x[i * 5 + 3];
  feat[i * 2 + 1] = x[i * 5 + 4];
}

// ---------------------------------------------------------------------------
// Farthest point sampling: one block per batch, min-dists in registers,
// block argmax (ties -> lowest index, matching jnp.argmax) via LDS tree.
// ---------------------------------------------------------------------------
template <int N, int M>
__global__ void __launch_bounds__(1024) fps_kernel(const float* __restrict__ pos,
                                                   int* __restrict__ idxOut) {
  constexpr int T = 1024;
  constexpr int PPT = N / T;
  __shared__ float sval[T];
  __shared__ int   sidx[T];
  __shared__ int   sLast;
  const int b = blockIdx.x;
  const int t = threadIdx.x;
  const float* P = pos + (long)b * N * 3;

  float mind[PPT];
#pragma unroll
  for (int j = 0; j < PPT; ++j) mind[j] = FBIG;
  if (t == 0) { idxOut[b * M] = 0; sLast = 0; }
  __syncthreads();

  for (int i = 1; i < M; ++i) {
    const int last = sLast;
    __syncthreads();
    const float lx = P[last * 3 + 0], ly = P[last * 3 + 1], lz = P[last * 3 + 2];
    float bestV = -1.f;
    int bestI = 0;
#pragma unroll
    for (int j = 0; j < PPT; ++j) {
      int p = t + j * T;
      float dx = P[p * 3 + 0] - lx;
      float dy = P[p * 3 + 1] - ly;
      float dz = P[p * 3 + 2] - lz;
      float d = dx * dx + dy * dy + dz * dz;
      float m = mind[j] < d ? mind[j] : d;
      mind[j] = m;
      if (m > bestV) { bestV = m; bestI = p; }  // ascending p -> lowest idx on tie
    }
    sval[t] = bestV;
    sidx[t] = bestI;
    __syncthreads();
    for (int s = T / 2; s > 0; s >>= 1) {
      if (t < s) {
        float v2 = sval[t + s];
        int i2 = sidx[t + s];
        if (v2 > sval[t] || (v2 == sval[t] && i2 < sidx[t])) { sval[t] = v2; sidx[t] = i2; }
      }
      __syncthreads();
    }
    if (t == 0) { sLast = sidx[0]; idxOut[b * M + i] = sidx[0]; }
    __syncthreads();
  }
}

// ---------------------------------------------------------------------------
// qpos = pos[idx]
// ---------------------------------------------------------------------------
__global__ void gather_q_kernel(const float* __restrict__ pos, const int* __restrict__ idx,
                                float* __restrict__ qpos, int Nin, int M) {
  int i = blockIdx.x * blockDim.x + threadIdx.x;
  int b = blockIdx.y;
  if (i >= M) return;
  int s = idx[b * M + i];
  const float* src = pos + ((long)b * Nin + s) * 3;
  float* dst = qpos + ((long)b * M + i) * 3;
  dst[0] = src[0]; dst[1] = src[1]; dst[2] = src[2];
}

// ---------------------------------------------------------------------------
// Radius top-K (K=32 smallest d2 within r2, ties -> lowest index == lax.top_k
// on -d2). One block per query; d2 cached in dynamic LDS; 32 argmin rounds.
// ---------------------------------------------------------------------------
__global__ void __launch_bounds__(256) knn_kernel(const float* __restrict__ pos,
                                                  const float* __restrict__ qpos,
                                                  int* __restrict__ nidx,
                                                  int* __restrict__ valid,
                                                  int Nin, int M, float r2) {
  extern __shared__ __align__(16) char dynsmem[];
  float* d2s = (float*)dynsmem;  // Nin floats
  __shared__ float rv[256];
  __shared__ int   ri[256];

  const int b = blockIdx.y;
  const int q = blockIdx.x;
  const int t = threadIdx.x;
  const float* P = pos + (long)b * Nin * 3;
  const float* Q = qpos + ((long)b * M + q) * 3;
  const float qx = Q[0], qy = Q[1], qz = Q[2];

  for (int p = t; p < Nin; p += 256) {
    float dx = P[p * 3 + 0] - qx;
    float dy = P[p * 3 + 1] - qy;
    float dz = P[p * 3 + 2] - qz;
    float d = dx * dx + dy * dy + dz * dz;
    d2s[p] = (d <= r2) ? d : 1e30f;
  }
  __syncthreads();

  for (int s = 0; s < 32; ++s) {
    float bv = 2e30f;
    int bi = 0;
    for (int p = t; p < Nin; p += 256) {
      float v = d2s[p];
      if (v < bv) { bv = v; bi = p; }  // ascending p -> lowest idx on tie
    }
    rv[t] = bv;
    ri[t] = bi;
    __syncthreads();
    for (int sr = 128; sr > 0; sr >>= 1) {
      if (t < sr) {
        float v2 = rv[t + sr];
        int i2 = ri[t + sr];
        if (v2 < rv[t] || (v2 == rv[t] && i2 < ri[t])) { rv[t] = v2; ri[t] = i2; }
      }
      __syncthreads();
    }
    if (t == 0) {
      long base = ((long)b * M + q) * 32 + s;
      nidx[base]  = ri[0];
      valid[base] = (rv[0] < 1e29f) ? 1 : 0;
      d2s[ri[0]] = 1e30f;
    }
    __syncthreads();
  }
}

// ---------------------------------------------------------------------------
// Grouped 3-layer MLP + masked max-pool via WMMA.
// Block = 128 threads (4 waves), one query per wave (32 neighbor-rows = 2
// M-tiles). Weights (transposed, f16) loaded into LDS once per block.
// ---------------------------------------------------------------------------
template <int CF, int CINP, int CH, int COUT>
__global__ void __launch_bounds__(128) mlp_pool_kernel(
    const float* __restrict__ feat, const float* __restrict__ pos,
    const float* __restrict__ qpos, const int* __restrict__ nidx,
    const int* __restrict__ valid,
    const float* __restrict__ w1, const float* __restrict__ b1,
    const float* __restrict__ w2, const float* __restrict__ b2,
    const float* __restrict__ w3, const float* __restrict__ b3,
    float* __restrict__ fout, int Nin, int M) {
  extern __shared__ __align__(16) char dynsmem[];
  constexpr int CIN = CF + 3;
  _Float16* wT1 = (_Float16*)dynsmem;              // CH x CINP
  float*    bs1 = (float*)(wT1 + CH * CINP);
  _Float16* wT2 = (_Float16*)(bs1 + CH);           // CH x CH
  float*    bs2 = (float*)(wT2 + CH * CH);
  _Float16* wT3 = (_Float16*)(bs2 + CH);           // COUT x CH
  float*    bs3 = (float*)(wT3 + COUT * CH);
  _Float16* act = (_Float16*)(bs3 + COUT);         // 4 waves x (32*CINP + 32*CH)

  const int tid  = threadIdx.x;
  const int lane = tid & 31;
  const int wave = tid >> 5;

  for (int i = tid; i < CH * CINP; i += 128) {
    int n = i / CINP, k = i % CINP;
    wT1[i] = (k < CIN) ? (_Float16)w1[k * CH + n] : (_Float16)0.f;
  }
  for (int i = tid; i < CH * CH; i += 128) {
    int n = i / CH, k = i % CH;
    wT2[i] = (_Float16)w2[k * CH + n];
  }
  for (int i = tid; i < COUT * CH; i += 128) {
    int n = i / CH, k = i % CH;
    wT3[i] = (_Float16)w3[k * COUT + n];
  }
  for (int i = tid; i < CH; i += 128) { bs1[i] = b1[i]; bs2[i] = b2[i]; }
  for (int i = tid; i < COUT; i += 128) bs3[i] = b3[i];
  __syncthreads();

  const int b = blockIdx.y;
  const int q = blockIdx.x * 4 + wave;
  _Float16* bufA = act + wave * (32 * CINP + 32 * CH);  // h0 / h2
  _Float16* bufB = bufA + 32 * CINP;                    // h1

  // stage 0: h0[j][:] = [feat[nidx[j]], pos[nidx[j]] - qpos, 0-pad]
  {
    const long base = ((long)b * M + q) * 32 + lane;
    int vld = valid[base];
    int nid = vld ? nidx[base] : 0;
    const float* fsrc = feat + ((long)b * Nin + nid) * CF;
    const float* psrc = pos + ((long)b * Nin + nid) * 3;
    const float* qp = qpos + ((long)b * M + q) * 3;
    _Float16* row = bufA + lane * CINP;
    for (int c = 0; c < CF; ++c) row[c] = (_Float16)fsrc[c];
    for (int c = 0; c < 3; ++c) row[CF + c] = (_Float16)(psrc[c] - qp[c]);
    for (int c = CIN; c < CINP; ++c) row[c] = (_Float16)0.f;
  }
  __syncthreads();

  dense_relu_layer<CINP, CH>(bufA, wT1, bs1, bufB, lane);   // h0 -> h1
  __syncthreads();
  dense_relu_layer<CH, CH>(bufB, wT2, bs2, bufA, lane);     // h1 -> h2
  __syncthreads();

  // layer 3 + masked max-pool over 32 neighbors
  {
    const int* vld = valid + ((long)b * M + q) * 32;
    const int n  = lane & 15;
    const int hs = lane >> 4;
    for (int nt = 0; nt < COUT / 16; ++nt) {
      float colmax = -FBIG;
      for (int tile = 0; tile < 2; ++tile) {
        v8f acc = {};
        for (int ks = 0; ks < CH / 32; ++ks) {
          v16h a = load_a_frag(bufA, CH, tile, ks * 32, lane);
          v16h w = load_b_frag(wT3, CH, nt, ks * 32, lane);
          acc = wmma_f16(a, w, acc);
        }
        float bb = bs3[nt * 16 + n];
#pragma unroll
        for (int v = 0; v < 8; ++v) {
          int r = tile * 16 + v + 8 * hs;  // neighbor row
          float val = acc[v] + bb;
          val = val > 0.f ? val : 0.f;
          if (!vld[r]) val = -FBIG;
          colmax = val > colmax ? val : colmax;
        }
      }
      float other = __shfl_xor(colmax, 16, 32);
      colmax = other > colmax ? other : colmax;
      if (lane < 16)
        fout[((long)b * M + q) * COUT + nt * 16 + n] = colmax;
    }
  }
}

// ---------------------------------------------------------------------------
static constexpr size_t mlp_lds_bytes(int CINP, int CH, int COUT) {
  return (size_t)CH * CINP * 2 + (size_t)CH * 4 + (size_t)CH * CH * 2 +
         (size_t)CH * 4 + (size_t)COUT * CH * 2 + (size_t)COUT * 4 +
         (size_t)4 * (32 * CINP + 32 * CH) * 2;
}

extern "C" void kernel_launch(void* const* d_in, const int* in_sizes, int n_in,
                              void* d_out, int out_size, void* d_ws, size_t ws_size,
                              hipStream_t stream) {
  (void)in_sizes; (void)n_in; (void)out_size; (void)ws_size;
  constexpr int B = 4, N = 8192, M1 = 4096, M2 = 1024;

  const float* x = (const float*)d_in[0];
  // d_in[1] = mask (unused by reference)
  const float* w1_0 = (const float*)d_in[2];
  const float* b1_0 = (const float*)d_in[3];
  const float* w1_1 = (const float*)d_in[4];
  const float* b1_1 = (const float*)d_in[5];
  const float* w1_2 = (const float*)d_in[6];
  const float* b1_2 = (const float*)d_in[7];
  const float* w2_0 = (const float*)d_in[8];
  const float* b2_0 = (const float*)d_in[9];
  const float* w2_1 = (const float*)d_in[10];
  const float* b2_1 = (const float*)d_in[11];
  const float* w2_2 = (const float*)d_in[12];
  const float* b2_2 = (const float*)d_in[13];

  float* out    = (float*)d_out;
  float* o_feat = out;                         // (B,N,2)
  float* o_pos  = o_feat + (size_t)B * N * 2;  // (B,N,3)
  float* o_f2   = o_pos + (size_t)B * N * 3;   // (B,M1,64)
  float* o_q2   = o_f2 + (size_t)B * M1 * 64;  // (B,M1,3)
  float* o_f3   = o_q2 + (size_t)B * M1 * 3;   // (B,M2,128)
  float* o_q3   = o_f3 + (size_t)B * M2 * 128; // (B,M2,3)

  int* ws     = (int*)d_ws;
  int* idx1   = ws;                     // B*M1
  int* idx2   = idx1 + B * M1;          // B*M2
  int* nidx1  = idx2 + B * M2;          // B*M1*32
  int* valid1 = nidx1 + B * M1 * 32;    // B*M1*32
  int* nidx2  = valid1 + B * M1 * 32;   // B*M2*32
  int* valid2 = nidx2 + B * M2 * 32;    // B*M2*32

  split_kernel<<<(B * N + 255) / 256, 256, 0, stream>>>(x, o_feat, o_pos, B * N);

  // ---- level 1: FPS(8192 -> 4096), r=0.5, MLP 5->32->32->64 ----
  fps_kernel<N, M1><<<B, 1024, 0, stream>>>(o_pos, idx1);
  gather_q_kernel<<<dim3((M1 + 255) / 256, B), 256, 0, stream>>>(o_pos, idx1, o_q2, N, M1);
  knn_kernel<<<dim3(M1, B), 256, (size_t)N * 4, stream>>>(o_pos, o_q2, nidx1, valid1,
                                                          N, M1, 0.25f);
  mlp_pool_kernel<2, 32, 32, 64>
      <<<dim3(M1 / 4, B), 128, mlp_lds_bytes(32, 32, 64), stream>>>(
          o_feat, o_pos, o_q2, nidx1, valid1,
          w1_0, b1_0, w1_1, b1_1, w1_2, b1_2, o_f2, N, M1);

  // ---- level 2: FPS(4096 -> 1024), r=1.0, MLP 67->64->64->128 ----
  fps_kernel<M1, M2><<<B, 1024, 0, stream>>>(o_q2, idx2);
  gather_q_kernel<<<dim3((M2 + 255) / 256, B), 256, 0, stream>>>(o_q2, idx2, o_q3, M1, M2);
  knn_kernel<<<dim3(M2, B), 256, (size_t)M1 * 4, stream>>>(o_q2, o_q3, nidx2, valid2,
                                                           M1, M2, 1.0f);
  mlp_pool_kernel<64, 96, 64, 128>
      <<<dim3(M2 / 4, B), 128, mlp_lds_bytes(96, 64, 128), stream>>>(
          o_f2, o_q2, o_q3, nidx2, valid2,
          w2_0, b2_0, w2_1, b2_1, w2_2, b2_2, o_f3, M1, M2);
}
